// EncoderRNN_11390253269480
// MI455X (gfx1250) — matrix operand order repaired and checked by reading, not compiled, not run on hardware
//
#include <hip/hip_runtime.h>
#include <hip/hip_bf16.h>
#include <stdint.h>

// ---------------------------------------------------------------------------
// EncoderRNN (embedding + 50-step GRU, B=128, H=512) for MI455X (gfx1250).
//
// Phase 1 (parallel):   x_bf16 = emb[tokens]           (gather + fp32->bf16)
//                       gi     = x @ w_ih^T + b_ih     (WMMA bf16 f32-acc,
//                                B staged via async-to-LDS, double buffered)
// Phase 2 (serial):     50-step recurrence, persistent kernel, w_hh packed
//                       in bf16 WMMA-B layout resident in LDS (loaded with
//                       async-to-LDS), one grid barrier per step,
//                       double-buffered hidden state.
// ---------------------------------------------------------------------------

#define VOCAB   50257
#define HIDDEN  512
#define BATCH   128
#define TSTEPS  50
#define G3      (3 * HIDDEN)          // 1536
#define HB      (BATCH * HIDDEN)      // 65536 elements of hidden state

typedef __attribute__((ext_vector_type(16))) __bf16         v16bf;
typedef __attribute__((ext_vector_type(8)))  float          v8f;
typedef __attribute__((ext_vector_type(8)))  unsigned short u16x8;
typedef __attribute__((ext_vector_type(16))) unsigned short u16x16;
typedef __attribute__((ext_vector_type(4)))  unsigned short u16x4;

union frag_u { u16x8 h[2]; u16x16 w; };

static __device__ __forceinline__ unsigned short f2bf(float f) {
    unsigned int u = __float_as_uint(f);
    u += 0x7FFFu + ((u >> 16) & 1u);          // round-to-nearest-even
    return (unsigned short)(u >> 16);
}

static __device__ __forceinline__ float sigmoidf_(float x) {
    return 1.0f / (1.0f + __expf(-x));
}

static __device__ __forceinline__ v8f wmma_bf16(v16bf a, v16bf b, v8f c) {
    return __builtin_amdgcn_wmma_f32_16x16x32_bf16(
        /*neg_a=*/false, a, /*neg_b=*/false, b,
        /*c_mod=*/(short)0, c, /*reuse_a=*/false, /*reuse_b=*/false);
}

// ---- CDNA5 async memory->LDS copy (ASYNCcnt-tracked), per ISA 15.18.3 ------
static __device__ __forceinline__ void async_ld_b128(uint32_t lds_off, uint64_t gaddr) {
    asm volatile("global_load_async_to_lds_b128 %0, %1, off"
                 :: "v"(lds_off), "v"(gaddr) : "memory");
}
#define WAIT_ASYNC(n) asm volatile("s_wait_asynccnt %0" :: "i"(n) : "memory")

static __device__ __forceinline__ uint32_t lds_addr32(const void* p) {
    // generic LDS pointers carry the byte offset in the low 32 bits
    return (uint32_t)(uintptr_t)p;
}

// ---------------------------------------------------------------------------
// Grid-wide sense barrier (persistent kernel). ctr[0]=count, ctr[1]=generation.
// ---------------------------------------------------------------------------
static __device__ __forceinline__ void grid_barrier(int* ctr, int nblk) {
    __syncthreads();
    if (threadIdx.x == 0) {
        __threadfence();
        int* cnt = ctr;
        int* gen = ctr + 1;
        int g = __hip_atomic_load(gen, __ATOMIC_ACQUIRE, __HIP_MEMORY_SCOPE_AGENT);
        int prev = __hip_atomic_fetch_add(cnt, 1, __ATOMIC_ACQ_REL, __HIP_MEMORY_SCOPE_AGENT);
        if (prev == nblk - 1) {
            __hip_atomic_store(cnt, 0, __ATOMIC_RELAXED, __HIP_MEMORY_SCOPE_AGENT);
            __hip_atomic_fetch_add(gen, 1, __ATOMIC_ACQ_REL, __HIP_MEMORY_SCOPE_AGENT);
        } else {
            while (__hip_atomic_load(gen, __ATOMIC_ACQUIRE, __HIP_MEMORY_SCOPE_AGENT) == g)
                __builtin_amdgcn_s_sleep(2);
        }
    }
    __syncthreads();
}

__global__ void init_ctr(int* ctr) { ctr[0] = 0; ctr[1] = 0; }

// ---------------------------------------------------------------------------
// Repack W [G rows, K=512 cols] fp32 -> bf16 WMMA B-fragment layout.
// Fragment (nt, kt) holds B[k][n] = W[nt*16+n][kt*32+k]; per-lane 16 contiguous
// bf16: n = lane%16, k = (i%8) + 8*(lane>=16) + 16*(i/8).  dst index:
//   ((nt*16 + kt)*32 + lane)*16 + i
// ---------------------------------------------------------------------------
__global__ __launch_bounds__(256) void repack_w(const float* __restrict__ W,
                                                unsigned short* __restrict__ dst,
                                                int nrows) {
    int idx = blockIdx.x * 256 + threadIdx.x;
    int total = nrows * HIDDEN;
    if (idx >= total) return;
    int frag = idx >> 9;            // /512
    int r    = idx & 511;
    int lane = r >> 4;
    int i    = r & 15;
    int nt   = frag >> 4;
    int kt   = frag & 15;
    int n    = lane & 15;
    int k    = (i & 7) + ((lane >> 4) << 3) + ((i >> 3) << 4);
    dst[idx] = f2bf(W[(size_t)(nt * 16 + n) * HIDDEN + kt * 32 + k]);
}

// ---------------------------------------------------------------------------
// Embedding gather + fp32 -> bf16.  x rows are [t*BATCH + b][HIDDEN].
// ---------------------------------------------------------------------------
__global__ __launch_bounds__(128) void embed_bf16(const int* __restrict__ tokens,
                                                  const float* __restrict__ emb,
                                                  unsigned short* __restrict__ xbf) {
    int row = blockIdx.x;
    int t = row / BATCH, b = row % BATCH;
    int tok = tokens[b * TSTEPS + t];     // input_tensor is [B, T]
    const float4* src = (const float4*)(emb + (size_t)tok * HIDDEN);
    float4 v = src[threadIdx.x];
    u16x4 o = { f2bf(v.x), f2bf(v.y), f2bf(v.z), f2bf(v.w) };
    ((u16x4*)(xbf + (size_t)row * HIDDEN))[threadIdx.x] = o;
}

// ---------------------------------------------------------------------------
// gi = x @ w_ih^T + b_ih.   M=6400, N=1536, K=512, bf16 WMMA f32-acc.
// Block = 256 threads (8 waves); block tile 128(M) x 64(N); wave tile 16x64.
// B slices (4 KB per k-step, shared by all 8 waves) staged via async-to-LDS,
// double-buffered; A fragments straight global->VGPR.
// ---------------------------------------------------------------------------
__global__ __launch_bounds__(256) void gi_gemm(const unsigned short* __restrict__ xbf,
                                               const unsigned short* __restrict__ wpack,
                                               const float* __restrict__ b_ih,
                                               float* __restrict__ gi) {
    __shared__ unsigned short bstage[2][4 * 512];   // 2 x 4 KB

    int bx = blockIdx.x;
    int mblk = bx % (TSTEPS * BATCH / 128);         // 50
    int nblk = bx / (TSTEPS * BATCH / 128);         // 24
    int tid  = threadIdx.x;
    int wave = tid >> 5;
    int lane = tid & 31;
    int lh   = lane >> 4;
    int l16  = lane & 15;
    int mbase = mblk * 128 + wave * 16;

    // staging: thread tid copies 16B chunk (nt = tid/64, c = tid%64) of the
    // current k-slice.  LDS layout is frag-linear: frag nt at bytes nt*1024.
    uint64_t bsrc = (uint64_t)(uintptr_t)(wpack
        + (size_t)(nblk * 4 + (tid >> 6)) * 16 * 512   // frag row base (nt)
        + (size_t)(tid & 63) * 8);                     // chunk within frag
    uint32_t ldst[2] = { lds_addr32(&bstage[0][tid * 8]),
                         lds_addr32(&bstage[1][tid * 8]) };

    v8f acc[4];
#pragma unroll
    for (int nt = 0; nt < 4; ++nt) acc[nt] = (v8f){0,0,0,0,0,0,0,0};

    const unsigned short* arow = xbf + (size_t)(mbase + l16) * HIDDEN;

    // prologue: stage k-slice 0
    async_ld_b128(ldst[0], bsrc);

    for (int kt = 0; kt < 16; ++kt) {
        int buf = kt & 1;
        WAIT_ASYNC(0);          // current k-slice landed in LDS
        __syncthreads();        // all waves past previous reads + see new data
        if (kt < 15)            // overlap next slice with this slice's math
            async_ld_b128(ldst[buf ^ 1], bsrc + (uint64_t)(kt + 1) * 1024);

        int k0 = kt * 32 + lh * 8;
        frag_u a;
        a.h[0] = *(const u16x8*)(arow + k0);
        a.h[1] = *(const u16x8*)(arow + k0 + 16);
        v16bf av = __builtin_bit_cast(v16bf, a.w);
#pragma unroll
        for (int nt = 0; nt < 4; ++nt) {
            u16x16 bw = *(const u16x16*)&bstage[buf][nt * 512 + lane * 16];
            v16bf bv = __builtin_bit_cast(v16bf, bw);
            acc[nt] = wmma_bf16(av, bv, acc[nt]);
        }
    }

#pragma unroll
    for (int nt = 0; nt < 4; ++nt) {
        int col  = (nblk * 4 + nt) * 16 + l16;
        float bias = b_ih[col];
#pragma unroll
        for (int v = 0; v < 8; ++v) {
            int row = mbase + v + lh * 8;
            gi[(size_t)row * G3 + col] = acc[nt][v] + bias;
        }
    }
}

// ---------------------------------------------------------------------------
// GRU recurrence.  64 persistent workgroups x 128 threads (4 waves).
// Workgroup (jt, mh): hidden-column slice j0=jt*16, batch half mh.
// w_hh B-fragments (48 KB) async-copied into LDS once; per step each wave
// computes r/z/n 16x16 tiles (K=512) with bf16 WMMA; h double-buffered.
// ---------------------------------------------------------------------------
#define NBLK 64
__global__ __launch_bounds__(128) void gru_step(const unsigned short* __restrict__ whh_pack,
                                                const float* __restrict__ b_hh,
                                                const float* __restrict__ gi,
                                                float* __restrict__ out,
                                                float* __restrict__ hf,
                                                unsigned short* __restrict__ hb,
                                                int* __restrict__ ctr) {
    int jt = blockIdx.x & 31;
    int mh = blockIdx.x >> 5;
    int j0 = jt * 16;
    int wave = threadIdx.x >> 5;
    int lane = threadIdx.x & 31;
    int lh = lane >> 4, l16 = lane & 15;
    int mt = mh * 4 + wave;

    __shared__ unsigned short ldsb[3 * 16 * 32 * 16];   // 48 KB: 3 gates x 16 kt

    // ---- async preload of w_hh fragments: 3 contiguous 16 KB gate blocks ----
    // gate g occupies LDS bytes [g*16384, +16384) = frags (g*32+jt)*16 .. +16
    for (int idx = threadIdx.x; idx < 3072; idx += 128) {   // 3072 x 16B chunks
        int g   = idx >> 10;                                 // /1024
        int rem = idx & 1023;
        uint64_t src = (uint64_t)(uintptr_t)(whh_pack
            + (size_t)(g * 32 + jt) * 16 * 512) + (uint64_t)rem * 16;
        async_ld_b128(lds_addr32(&ldsb[(size_t)idx * 8]), src);
    }
    WAIT_ASYNC(0);

    float bhh_r = b_hh[j0 + l16];
    float bhh_z = b_hh[HIDDEN + j0 + l16];
    float bhh_n = b_hh[2 * HIDDEN + j0 + l16];

    // ---- zero-init h (buffer 0) ----
    for (int i = threadIdx.x + blockIdx.x * 128; i < HB; i += 128 * NBLK) {
        hf[i] = 0.0f;
        hb[i] = 0;
    }
    __syncthreads();
    grid_barrier(ctr, NBLK);

    int jcol = j0 + l16;

    for (int t = 0; t < TSTEPS; ++t) {
        int cur = t & 1, nxt = cur ^ 1;
        const unsigned short* hcur = hb + (size_t)cur * HB;
        const float*          hfc  = hf + (size_t)cur * HB;

        __builtin_prefetch(gi + ((size_t)t * BATCH + mt * 16 + l16) * G3 + j0, 0, 1);

        v8f accR = (v8f){0,0,0,0,0,0,0,0};
        v8f accZ = (v8f){0,0,0,0,0,0,0,0};
        v8f accN = (v8f){0,0,0,0,0,0,0,0};

        const unsigned short* arow = hcur + (size_t)(mt * 16 + l16) * HIDDEN;

        // manually double-buffered A fragments: load kt+1 before consuming kt
        frag_u a_cur;
        a_cur.h[0] = *(const u16x8*)(arow + lh * 8);
        a_cur.h[1] = *(const u16x8*)(arow + lh * 8 + 16);
#pragma unroll 4
        for (int kt = 0; kt < 16; ++kt) {
            frag_u a_nxt;
            if (kt < 15) {
                int k0 = (kt + 1) * 32 + lh * 8;
                a_nxt.h[0] = *(const u16x8*)(arow + k0);
                a_nxt.h[1] = *(const u16x8*)(arow + k0 + 16);
            }
            v16bf av = __builtin_bit_cast(v16bf, a_cur.w);
            u16x16 b0 = *(const u16x16*)&ldsb[((0 * 16 + kt) * 32 + lane) * 16];
            u16x16 b1 = *(const u16x16*)&ldsb[((1 * 16 + kt) * 32 + lane) * 16];
            u16x16 b2 = *(const u16x16*)&ldsb[((2 * 16 + kt) * 32 + lane) * 16];
            accR = wmma_bf16(av, __builtin_bit_cast(v16bf, b0), accR);
            accZ = wmma_bf16(av, __builtin_bit_cast(v16bf, b1), accZ);
            accN = wmma_bf16(av, __builtin_bit_cast(v16bf, b2), accN);
            a_cur = a_nxt;
        }

        // ---- gate nonlinearity + h update (C-layout: m = v + 8*lh, n = l16)
#pragma unroll
        for (int v = 0; v < 8; ++v) {
            int m = mt * 16 + v + lh * 8;
            size_t gbase = ((size_t)t * BATCH + m) * G3;
            float gr = gi[gbase + jcol];
            float gz = gi[gbase + HIDDEN + jcol];
            float gn = gi[gbase + 2 * HIDDEN + jcol];
            float hp = hfc[(size_t)m * HIDDEN + jcol];
            float r  = sigmoidf_(gr + accR[v] + bhh_r);
            float z  = sigmoidf_(gz + accZ[v] + bhh_z);
            float nn = tanhf(gn + r * (accN[v] + bhh_n));
            float hn = (1.0f - z) * nn + z * hp;
            size_t oix = (size_t)m * HIDDEN + jcol;
            out[(size_t)t * HB + oix] = hn;
            hf[(size_t)nxt * HB + oix] = hn;
            hb[(size_t)nxt * HB + oix] = f2bf(hn);
            if (t == TSTEPS - 1)
                out[(size_t)TSTEPS * HB + oix] = hn;   // encoder_hidden
        }
        grid_barrier(ctr, NBLK);
    }
}

// ---------------------------------------------------------------------------
extern "C" void kernel_launch(void* const* d_in, const int* in_sizes, int n_in,
                              void* d_out, int out_size, void* d_ws, size_t ws_size,
                              hipStream_t stream) {
    const int*   tokens = (const int*)d_in[0];
    const float* emb    = (const float*)d_in[1];
    const float* w_ih   = (const float*)d_in[2];
    const float* w_hh   = (const float*)d_in[3];
    const float* b_ih   = (const float*)d_in[4];
    const float* b_hh   = (const float*)d_in[5];
    float* out = (float*)d_out;

    char* ws = (char*)d_ws;
    size_t off = 0;
    auto carve = [&](size_t bytes) -> char* {
        char* p = ws + off;
        off = (off + bytes + 255) & ~(size_t)255;
        return p;
    };
    int*            ctr       = (int*)carve(2 * sizeof(int));
    unsigned short* wih_pack  = (unsigned short*)carve((size_t)G3 * HIDDEN * 2);
    unsigned short* whh_pack  = (unsigned short*)carve((size_t)G3 * HIDDEN * 2);
    unsigned short* xbf       = (unsigned short*)carve((size_t)TSTEPS * BATCH * HIDDEN * 2);
    float*          gi        = (float*)carve((size_t)TSTEPS * BATCH * G3 * 4);
    float*          hf        = (float*)carve((size_t)2 * HB * 4);
    unsigned short* hb        = (unsigned short*)carve((size_t)2 * HB * 2);
    (void)ws_size; (void)in_sizes; (void)n_in; (void)out_size;

    init_ctr<<<1, 1, 0, stream>>>(ctr);

    int pack_elems = G3 * HIDDEN;
    repack_w<<<(pack_elems + 255) / 256, 256, 0, stream>>>(w_ih, wih_pack, G3);
    repack_w<<<(pack_elems + 255) / 256, 256, 0, stream>>>(w_hh, whh_pack, G3);

    embed_bf16<<<TSTEPS * BATCH, 128, 0, stream>>>(tokens, emb, xbf);

    gi_gemm<<<(TSTEPS * BATCH / 128) * (G3 / 64), 256, 0, stream>>>(xbf, wih_pack, b_ih, gi);

    gru_step<<<NBLK, 128, 0, stream>>>(whh_pack, b_hh, gi, out, hf, hb, ctr);
}
